// SSAMBAEncoder_32590211842336
// MI455X (gfx1250) — matrix operand order repaired and checked by reading, not compiled, not run on hardware
//
#include <hip/hip_runtime.h>
#include <cstdint>
#include <cstddef>

// ---------------------------------------------------------------------------
// SSAMBA (bidirectional Mamba vision encoder) for MI455X / gfx1250.
// GEMMs run on v_wmma_f32_16x16x32_bf16 (wave32 WMMA); tiles staged to LDS via
// gfx1250 async global->LDS DMA (global_load_async_to_lds_b128, ASYNCcnt).
// Scan stays f32 VALU (latency-bound, 16 exp/step per channel).
// ---------------------------------------------------------------------------

typedef __bf16 bf16_t;
typedef __attribute__((ext_vector_type(16))) __bf16 v16bf;
typedef __attribute__((ext_vector_type(8)))  float  v8f;
typedef int v4i_vs __attribute__((vector_size(16)));   // matches builtin's V4i

#define AS1 __attribute__((address_space(1)))
#define AS3 __attribute__((address_space(3)))

#if defined(__has_builtin)
#if __has_builtin(__builtin_amdgcn_global_load_async_to_lds_b128) && \
    __has_builtin(__builtin_amdgcn_s_wait_asynccnt)
#define HAVE_ASYNC_LDS 1
#endif
#endif
#ifndef HAVE_ASYNC_LDS
#define HAVE_ASYNC_LDS 0
#endif

#if HAVE_ASYNC_LDS
// global -> LDS async 16B copy (per active lane), tracked by ASYNCcnt
#define GLD_ASYNC_B128(glb_src, lds_dst)                                   \
    __builtin_amdgcn_global_load_async_to_lds_b128(                        \
        (AS1 v4i_vs*)(void*)(glb_src), (AS3 v4i_vs*)(void*)(lds_dst), 0, 0)
#endif

#define DEPTH    24
#define D_MODEL  768
#define D_INNER  1536
#define D_STATE  16
#define DT_RANK  48
#define LTOK     513
#define BATCH    4
#define MROWS    (BATCH * LTOK)      // 2052 token rows
#define NPATCH   512
#define DBL_N    (DT_RANK + 2 * D_STATE)  // 80

// ======================= bf16 WMMA GEMM: C = A * B^T =======================
// A: MxK (row-major, lda), B: NxK (row-major, ldb; weight [out,in]),
// C: MxN f32 (ldc); Cb optional bf16 copy of C.  Requires K % 8 == 0.
#define BM 128
#define BN 64
#define BK 32

__global__ __launch_bounds__(256)
void gemm_bf16_nt(const bf16_t* __restrict__ A, int lda,
                  const bf16_t* __restrict__ B, int ldb,
                  float* __restrict__ C, bf16_t* __restrict__ Cb, int ldc,
                  int M, int N, int K)
{
    __shared__ bf16_t As[BM][BK];
    __shared__ bf16_t Bs[BN][BK];

    const int tid  = threadIdx.x;
    const int wave = tid >> 5;
    const int lane = tid & 31;
    const int half = lane >> 4;
    const int l16  = lane & 15;
    const int m0   = blockIdx.y * BM;
    const int n0   = blockIdx.x * BN;

    v8f acc0 = {}, acc1 = {}, acc2 = {}, acc3 = {};

    for (int k0 = 0; k0 < K; k0 += BK) {
        // ---- stage A tile (128x32): 512 chunks of 8 bf16 (16B each) ----
        #pragma unroll
        for (int i = tid; i < (BM * BK / 8); i += 256) {
            int r  = i >> 2;
            int kc = (i & 3) << 3;
            int gm = m0 + r, gk = k0 + kc;
            bool inb = (gm < M) && (gk < K);     // K%8==0: chunk all-in or all-out
#if HAVE_ASYNC_LDS
            if (inb) {
                GLD_ASYNC_B128(A + (size_t)gm * lda + gk, &As[r][kc]);
            } else {
                *(uint4*)&As[r][kc] = make_uint4(0u, 0u, 0u, 0u);
            }
#else
            uint4 t = make_uint4(0u, 0u, 0u, 0u);
            if (inb) {
                t = *(const uint4*)(A + (size_t)gm * lda + gk);
                if (gk + BK < K)   // gfx1250 global_prefetch_b8 of next K tile
                    __builtin_prefetch(A + (size_t)gm * lda + gk + BK, 0, 1);
            }
            *(uint4*)&As[r][kc] = t;
#endif
        }
        // ---- stage B tile (64x32): 256 chunks, one per thread ----
        {
            int r  = tid >> 2;
            int kc = (tid & 3) << 3;
            int gn = n0 + r, gk = k0 + kc;
            bool inb = (gn < N) && (gk < K);
#if HAVE_ASYNC_LDS
            if (inb) {
                GLD_ASYNC_B128(B + (size_t)gn * ldb + gk, &Bs[r][kc]);
            } else {
                *(uint4*)&Bs[r][kc] = make_uint4(0u, 0u, 0u, 0u);
            }
#else
            uint4 t = make_uint4(0u, 0u, 0u, 0u);
            if (inb) t = *(const uint4*)(B + (size_t)gn * ldb + gk);
            *(uint4*)&Bs[r][kc] = t;
#endif
        }
#if HAVE_ASYNC_LDS
        __builtin_amdgcn_s_wait_asynccnt(0);
#endif
        __syncthreads();

        // ---- A fragment (16x32, ISA 16-bit A layout) ----
        union { uint4 u[2]; v16bf v; } af;
        const int mr = wave * 16 + l16;
        af.u[0] = *(const uint4*)&As[mr][half * 8];       // K = 8*half .. +7
        af.u[1] = *(const uint4*)&As[mr][16 + half * 8];  // K = 16+8*half .. +7

        // ---- 4 B fragments (32x16 each) + WMMA ----
        union BF { uint4 u[2]; v16bf v; };
        #define DO_NT(nt, accv)                                                    \
        {                                                                          \
            BF bfr;                                                                \
            bfr.u[0] = *(const uint4*)&Bs[(nt)*16 + l16][half * 8];                \
            bfr.u[1] = *(const uint4*)&Bs[(nt)*16 + l16][16 + half * 8];           \
            accv = __builtin_amdgcn_wmma_f32_16x16x32_bf16(                        \
                false, af.v, false, bfr.v, (short)0, accv, false, false);          \
        }
        DO_NT(0, acc0)
        DO_NT(1, acc1)
        DO_NT(2, acc2)
        DO_NT(3, acc3)
        #undef DO_NT
        __syncthreads();
    }

    // ---- epilogue: C/D layout (VGPR v: M = v + 8*half; N = lane&15) ----
    const int mbase = m0 + wave * 16 + half * 8;
    v8f accs[4] = {acc0, acc1, acc2, acc3};
    #pragma unroll
    for (int nt = 0; nt < 4; ++nt) {
        int gn = n0 + nt * 16 + l16;
        if (gn >= N) continue;
        #pragma unroll
        for (int v = 0; v < 8; ++v) {
            int gm = mbase + v;
            if (gm < M) {
                float val = accs[nt][v];
                C[(size_t)gm * ldc + gn] = val;
                if (Cb) Cb[(size_t)gm * ldc + gn] = (bf16_t)val;
            }
        }
    }
}

// ============================ helper kernels ===============================

__global__ void k_f32_to_bf16(const float* __restrict__ s, bf16_t* __restrict__ d, long n)
{
    long i = (long)blockIdx.x * blockDim.x + threadIdx.x;
    if (i < n) d[i] = (bf16_t)s[i];
}

// x(4,128,1024) -> xp(4, 512, 256) bf16 patches
__global__ void k_build_patches(const float* __restrict__ x, bf16_t* __restrict__ xp)
{
    long i = (long)blockIdx.x * blockDim.x + threadIdx.x;
    const long total = (long)BATCH * NPATCH * 256;
    if (i >= total) return;
    int pp  = (int)(i % 256);
    long r  = i / 256;
    int n   = (int)(r % NPATCH);
    int b   = (int)(r / NPATCH);
    int f = n / 64, t = n % 64;
    int pr = pp / 16, pc = pp % 16;
    float v = x[((long)b * 128 + f * 16 + pr) * 1024 + t * 16 + pc];
    xp[i] = (bf16_t)v;
}

// tok assembly: h[r][d] = (cls | patchGEMM + patch_b) + pos ; res = 0
__global__ void k_assemble_tok(const float* __restrict__ tokg,
                               const float* __restrict__ patch_b,
                               const float* __restrict__ cls,
                               const float* __restrict__ pos,
                               float* __restrict__ h, float* __restrict__ res)
{
    long i = (long)blockIdx.x * blockDim.x + threadIdx.x;
    const long total = (long)MROWS * D_MODEL;
    if (i >= total) return;
    int d = (int)(i % D_MODEL);
    int r = (int)(i / D_MODEL);
    int b = r / LTOK, t = r % LTOK;
    float v = pos[(long)t * D_MODEL + d];
    if (t == 0) v += cls[d];
    else        v += tokg[((long)b * NPATCH + (t - 1)) * D_MODEL + d] + patch_b[d];
    h[i]   = v;
    res[i] = 0.0f;
}

// res += h ; hn = bf16(rms_norm(res, w))   (one block per token row)
__global__ __launch_bounds__(256)
void k_prenorm(float* __restrict__ res, const float* __restrict__ h,
               bf16_t* __restrict__ hn, const float* __restrict__ w)
{
    __shared__ float red[9];
    const int r = blockIdx.x;
    const size_t base = (size_t)r * D_MODEL;
    float loc[3]; float ss = 0.f;
    #pragma unroll
    for (int i = 0; i < 3; ++i) {
        int d = threadIdx.x + i * 256;
        float v = res[base + d] + h[base + d];
        res[base + d] = v;
        loc[i] = v;
        ss += v * v;
    }
    for (int o = 16; o > 0; o >>= 1) ss += __shfl_down(ss, o, 32);
    if ((threadIdx.x & 31) == 0) red[threadIdx.x >> 5] = ss;
    __syncthreads();
    if (threadIdx.x < 32) {
        float t = (threadIdx.x < 8) ? red[threadIdx.x] : 0.f;
        for (int o = 4; o > 0; o >>= 1) t += __shfl_down(t, o, 32);
        if (threadIdx.x == 0) red[8] = t;
    }
    __syncthreads();
    float rs = rsqrtf(red[8] / (float)D_MODEL + 1e-5f);
    #pragma unroll
    for (int i = 0; i < 3; ++i) {
        int d = threadIdx.x + i * 256;
        hn[base + d] = (bf16_t)(loc[i] * rs * w[d]);
    }
}

// depthwise causal conv(K=4) + SiLU, both branches (branch 1 time-reversed)
__global__ void k_conv_silu(const float* __restrict__ xz,
                            const float* __restrict__ conv_w,  // [2][1536][4] (layer)
                            const float* __restrict__ conv_b,  // [2][1536]
                            float* __restrict__ cf, bf16_t* __restrict__ cb)
{
    long i = (long)blockIdx.x * blockDim.x + threadIdx.x;
    const long per = (long)MROWS * D_INNER;
    if (i >= 2 * per) return;
    int br = (int)(i / per);
    long rem = i % per;
    int d = (int)(rem % D_INNER);
    int r = (int)(rem / D_INNER);
    int b = r / LTOK, t = r % LTOK;
    const float* wv = conv_w + ((long)br * D_INNER + d) * 4;
    float acc = conv_b[(long)br * D_INNER + d];
    #pragma unroll
    for (int k = 0; k < 4; ++k) {
        int tt = t - 3 + k;
        if (tt >= 0) {
            int nat = br ? (LTOK - 1 - tt) : tt;
            acc += wv[k] * xz[((size_t)b * LTOK + nat) * (2 * D_INNER) + d];
        }
    }
    float s = acc / (1.f + __expf(-acc));   // SiLU
    cf[i] = s;
    cb[i] = (bf16_t)s;
}

// selective scan: one thread per (branch, batch, channel); 513 sequential steps
__global__ __launch_bounds__(256)
void k_ssm_scan(const float* __restrict__ dtpre,  // [2][MROWS][1536]
                const float* __restrict__ bdt,    // [2][1536]
                const float* __restrict__ u,      // [2][MROWS][1536]  (conv out)
                const float* __restrict__ dbl,    // [2][MROWS][80]
                const float* __restrict__ a_log,  // [2][1536][16]
                const float* __restrict__ dd,     // [2][1536]
                const float* __restrict__ xz,     // [MROWS][3072]  (z = cols 1536..)
                float* __restrict__ y0, float* __restrict__ y1)
{
    int idx = blockIdx.x * blockDim.x + threadIdx.x;
    if (idx >= 2 * BATCH * D_INNER) return;
    int br  = idx / (BATCH * D_INNER);
    int rem = idx % (BATCH * D_INNER);
    int b   = rem / D_INNER;
    int d   = rem % D_INNER;

    float a[D_STATE];
    #pragma unroll
    for (int s = 0; s < D_STATE; ++s)
        a[s] = -__expf(a_log[((long)br * D_INNER + d) * D_STATE + s]);
    const float Dv = dd[(long)br * D_INNER + d];
    const float bd = bdt[(long)br * D_INNER + d];

    float hs[D_STATE];
    #pragma unroll
    for (int s = 0; s < D_STATE; ++s) hs[s] = 0.f;

    float* y = br ? y1 : y0;
    const float* dtp = dtpre + (size_t)br * MROWS * D_INNER;
    const float* up  = u     + (size_t)br * MROWS * D_INNER;
    const float* db  = dbl   + (size_t)br * MROWS * DBL_N;

    for (int t = 0; t < LTOK; ++t) {
        int rl = b * LTOK + t;
        float dt = dtp[(size_t)rl * D_INNER + d] + bd;
        dt = (dt > 20.f) ? dt : log1pf(__expf(dt));   // softplus
        float uu = up[(size_t)rl * D_INNER + d];
        float du = dt * uu;
        const float* Bt = db + (size_t)rl * DBL_N + DT_RANK;
        const float* Ct = Bt + D_STATE;
        float yv = 0.f;
        #pragma unroll
        for (int s = 0; s < D_STATE; ++s) {
            float dA = __expf(dt * a[s]);
            hs[s] = dA * hs[s] + du * Bt[s];
            yv += hs[s] * Ct[s];
        }
        yv += uu * Dv;
        int nat = br ? (LTOK - 1 - t) : t;
        float z = xz[((size_t)b * LTOK + nat) * (2 * D_INNER) + D_INNER + d];
        yv *= z / (1.f + __expf(-z));
        y[((size_t)b * LTOK + nat) * D_INNER + d] = yv;
    }
}

// y_b = bf16(0.5*(y0+y1))
__global__ void k_combine(const float* __restrict__ y0, const float* __restrict__ y1,
                          bf16_t* __restrict__ yb)
{
    long i = (long)blockIdx.x * blockDim.x + threadIdx.x;
    const long total = (long)MROWS * D_INNER;
    if (i >= total) return;
    yb[i] = (bf16_t)(0.5f * (y0[i] + y1[i]));
}

// final: layer_norm(res[:,0]+h[:,0]) -> out (one block per batch)
__global__ __launch_bounds__(256)
void k_final_ln(const float* __restrict__ res, const float* __restrict__ h,
                const float* __restrict__ w, const float* __restrict__ bb,
                float* __restrict__ out)
{
    __shared__ float red[9];
    __shared__ float stat[2];
    const int bidx = blockIdx.x;
    const size_t base = (size_t)bidx * LTOK * D_MODEL;
    float v[3]; float s = 0.f;
    #pragma unroll
    for (int i = 0; i < 3; ++i) {
        int d = threadIdx.x + i * 256;
        float t = res[base + d] + h[base + d];
        v[i] = t; s += t;
    }
    for (int o = 16; o > 0; o >>= 1) s += __shfl_down(s, o, 32);
    if ((threadIdx.x & 31) == 0) red[threadIdx.x >> 5] = s;
    __syncthreads();
    if (threadIdx.x < 32) {
        float t = (threadIdx.x < 8) ? red[threadIdx.x] : 0.f;
        for (int o = 4; o > 0; o >>= 1) t += __shfl_down(t, o, 32);
        if (threadIdx.x == 0) stat[0] = t / (float)D_MODEL;
    }
    __syncthreads();
    float mean = stat[0];
    __syncthreads();
    float s2 = 0.f;
    #pragma unroll
    for (int i = 0; i < 3; ++i) { float dv = v[i] - mean; s2 += dv * dv; }
    for (int o = 16; o > 0; o >>= 1) s2 += __shfl_down(s2, o, 32);
    if ((threadIdx.x & 31) == 0) red[threadIdx.x >> 5] = s2;
    __syncthreads();
    if (threadIdx.x < 32) {
        float t = (threadIdx.x < 8) ? red[threadIdx.x] : 0.f;
        for (int o = 4; o > 0; o >>= 1) t += __shfl_down(t, o, 32);
        if (threadIdx.x == 0) stat[1] = rsqrtf(t / (float)D_MODEL + 1e-5f);
    }
    __syncthreads();
    float rstd = stat[1];
    #pragma unroll
    for (int i = 0; i < 3; ++i) {
        int d = threadIdx.x + i * 256;
        out[(size_t)bidx * D_MODEL + d] = (v[i] - mean) * rstd * w[d] + bb[d];
    }
}

// ============================== launch =====================================

static inline unsigned cdiv(long a, long b) { return (unsigned)((a + b - 1) / b); }

extern "C" void kernel_launch(void* const* d_in, const int* in_sizes, int n_in,
                              void* d_out, int out_size, void* d_ws, size_t ws_size,
                              hipStream_t stream)
{
    (void)in_sizes; (void)n_in; (void)out_size; (void)ws_size;

    const float* x         = (const float*)d_in[0];
    const float* patch_w   = (const float*)d_in[1];
    const float* patch_b   = (const float*)d_in[2];
    const float* cls_token = (const float*)d_in[3];
    const float* pos_embed = (const float*)d_in[4];
    const float* ln_w      = (const float*)d_in[5];
    const float* w_in      = (const float*)d_in[6];
    const float* conv_w    = (const float*)d_in[7];
    const float* conv_b    = (const float*)d_in[8];
    const float* w_x       = (const float*)d_in[9];
    const float* w_dt      = (const float*)d_in[10];
    const float* b_dt      = (const float*)d_in[11];
    const float* a_log     = (const float*)d_in[12];
    const float* dd        = (const float*)d_in[13];
    const float* w_out     = (const float*)d_in[14];
    const float* fn_w      = (const float*)d_in[15];
    const float* fn_b      = (const float*)d_in[16];
    float* out = (float*)d_out;

    // ---- workspace carve-out (256B aligned) ----
    size_t off = 0;
    auto wsalloc = [&](size_t nbytes) -> char* {
        char* p = (char*)d_ws + off;
        off += (nbytes + 255) & ~(size_t)255;
        return p;
    };
    const long n_w_in  = (long)DEPTH * 2 * D_INNER * D_MODEL;      // 24*3072*768
    const long n_w_x   = (long)DEPTH * 2 * DBL_N * D_INNER;
    const long n_w_dt  = (long)DEPTH * 2 * D_INNER * DT_RANK;
    const long n_w_out = (long)DEPTH * D_MODEL * D_INNER;
    const long n_pw    = (long)D_MODEL * 256;

    bf16_t* w_in_b  = (bf16_t*)wsalloc((size_t)n_w_in  * 2);
    bf16_t* w_x_b   = (bf16_t*)wsalloc((size_t)n_w_x   * 2);
    bf16_t* w_dt_b  = (bf16_t*)wsalloc((size_t)n_w_dt  * 2);
    bf16_t* w_out_b = (bf16_t*)wsalloc((size_t)n_w_out * 2);
    bf16_t* pw_b    = (bf16_t*)wsalloc((size_t)n_pw    * 2);
    bf16_t* xp_b    = (bf16_t*)wsalloc((size_t)BATCH * NPATCH * 256 * 2);
    float*  tokg    = (float*) wsalloc((size_t)BATCH * NPATCH * D_MODEL * 4);
    float*  hbuf    = (float*) wsalloc((size_t)MROWS * D_MODEL * 4);
    float*  resbuf  = (float*) wsalloc((size_t)MROWS * D_MODEL * 4);
    bf16_t* hn_b    = (bf16_t*)wsalloc((size_t)MROWS * D_MODEL * 2);
    float*  xz      = (float*) wsalloc((size_t)MROWS * 2 * D_INNER * 4);
    float*  cf      = (float*) wsalloc((size_t)2 * MROWS * D_INNER * 4);
    bf16_t* cb      = (bf16_t*)wsalloc((size_t)2 * MROWS * D_INNER * 2);
    float*  dblf    = (float*) wsalloc((size_t)2 * MROWS * DBL_N * 4);
    bf16_t* dbl_b   = (bf16_t*)wsalloc((size_t)2 * MROWS * DBL_N * 2);
    float*  dtpre   = (float*) wsalloc((size_t)2 * MROWS * D_INNER * 4);
    float*  y0      = (float*) wsalloc((size_t)MROWS * D_INNER * 4);
    float*  y1      = (float*) wsalloc((size_t)MROWS * D_INNER * 4);
    bf16_t* y_b     = (bf16_t*)wsalloc((size_t)MROWS * D_INNER * 2);

    // ---- weight conversion to bf16 (once per call) ----
    k_f32_to_bf16<<<cdiv(n_w_in, 256),  256, 0, stream>>>(w_in,  w_in_b,  n_w_in);
    k_f32_to_bf16<<<cdiv(n_w_x, 256),   256, 0, stream>>>(w_x,   w_x_b,   n_w_x);
    k_f32_to_bf16<<<cdiv(n_w_dt, 256),  256, 0, stream>>>(w_dt,  w_dt_b,  n_w_dt);
    k_f32_to_bf16<<<cdiv(n_w_out, 256), 256, 0, stream>>>(w_out, w_out_b, n_w_out);
    k_f32_to_bf16<<<cdiv(n_pw, 256),    256, 0, stream>>>(patch_w, pw_b,  n_pw);

    // ---- patch embedding ----
    k_build_patches<<<cdiv((long)BATCH * NPATCH * 256, 256), 256, 0, stream>>>(x, xp_b);
    {
        dim3 g(cdiv(D_MODEL, BN), cdiv(BATCH * NPATCH, BM));
        gemm_bf16_nt<<<g, 256, 0, stream>>>(xp_b, 256, pw_b, 256,
                                            tokg, nullptr, D_MODEL,
                                            BATCH * NPATCH, D_MODEL, 256);
    }
    k_assemble_tok<<<cdiv((long)MROWS * D_MODEL, 256), 256, 0, stream>>>(
        tokg, patch_b, cls_token, pos_embed, hbuf, resbuf);

    // ---- layer loop ----
    for (int l = 0; l < DEPTH; ++l) {
        k_prenorm<<<MROWS, 256, 0, stream>>>(resbuf, hbuf, hn_b, ln_w + (long)l * D_MODEL);

        { // xz = hn @ w_in^T  (2052 x 3072)
            dim3 g(cdiv(2 * D_INNER, BN), cdiv(MROWS, BM));
            gemm_bf16_nt<<<g, 256, 0, stream>>>(
                hn_b, D_MODEL, w_in_b + (size_t)l * 2 * D_INNER * D_MODEL, D_MODEL,
                xz, nullptr, 2 * D_INNER, MROWS, 2 * D_INNER, D_MODEL);
        }

        k_conv_silu<<<cdiv((long)2 * MROWS * D_INNER, 256), 256, 0, stream>>>(
            xz, conv_w + (long)l * 2 * D_INNER * 4, conv_b + (long)l * 2 * D_INNER, cf, cb);

        for (int br = 0; br < 2; ++br) { // dbl = c @ w_x^T  (2052 x 80)
            dim3 g(cdiv(DBL_N, BN), cdiv(MROWS, BM));
            gemm_bf16_nt<<<g, 256, 0, stream>>>(
                cb + (size_t)br * MROWS * D_INNER, D_INNER,
                w_x_b + ((size_t)l * 2 + br) * DBL_N * D_INNER, D_INNER,
                dblf + (size_t)br * MROWS * DBL_N,
                dbl_b + (size_t)br * MROWS * DBL_N, DBL_N,
                MROWS, DBL_N, D_INNER);
        }

        for (int br = 0; br < 2; ++br) { // dtpre = dbl[:, :48] @ w_dt^T  (2052 x 1536)
            dim3 g(cdiv(D_INNER, BN), cdiv(MROWS, BM));
            gemm_bf16_nt<<<g, 256, 0, stream>>>(
                dbl_b + (size_t)br * MROWS * DBL_N, DBL_N,
                w_dt_b + ((size_t)l * 2 + br) * D_INNER * DT_RANK, DT_RANK,
                dtpre + (size_t)br * MROWS * D_INNER, nullptr, D_INNER,
                MROWS, D_INNER, DT_RANK);
        }

        k_ssm_scan<<<cdiv(2 * BATCH * D_INNER, 256), 256, 0, stream>>>(
            dtpre, b_dt + (long)l * 2 * D_INNER, cf, dblf,
            a_log + (long)l * 2 * D_INNER * D_STATE, dd + (long)l * 2 * D_INNER,
            xz, y0, y1);

        k_combine<<<cdiv((long)MROWS * D_INNER, 256), 256, 0, stream>>>(y0, y1, y_b);

        { // h = y @ w_out^T  (2052 x 768)
            dim3 g(cdiv(D_MODEL, BN), cdiv(MROWS, BM));
            gemm_bf16_nt<<<g, 256, 0, stream>>>(
                y_b, D_INNER, w_out_b + (size_t)l * D_MODEL * D_INNER, D_INNER,
                hbuf, nullptr, D_MODEL, MROWS, D_MODEL, D_INNER);
        }
    }

    k_final_ln<<<BATCH, 256, 0, stream>>>(resbuf, hbuf, fn_w, fn_b, out);
}
